// SO3Convolution_9775345565773
// MI455X (gfx1250) — matrix operand order repaired and checked by reading, not compiled, not run on hardware
//
#include <hip/hip_runtime.h>
#include <hip/hip_bf16.h>
#include <cmath>

typedef __attribute__((ext_vector_type(16))) __bf16 v16bf;
typedef __attribute__((ext_vector_type(8)))  float  v8f;

#define RADIAL_DIM 12
#define NHID 288          // WEIGHT_NUMEL/2
#define NW   576          // WEIGHT_NUMEL
#define NT1  18           // NHID/16 n-tiles for GEMM1
#define NT2  36           // NW/16   n-tiles for GEMM2
#define NKB  9            // NHID/32 k-blocks for GEMM2
#define H_STRIDE 292      // padded bf16 row stride (146 dw -> 16 rows distinct banks)
#define XFEAT 40          // MUL0 + 3*MUL1
#define PANEL_BYTES 9216  // one W2 n-panel: 9 kb * 32 lanes * 32 B
#define H_BYTES (16 * H_STRIDE * 2)          // 9344 B per-wave GEMM1 output
#define REGION_A (4 * H_BYTES)               // 37376 B: h[4] overlaid with 2 panels
#define VEC_BYTES 4352                        // per-wave epilogue vectors
#define SHM_BYTES (REGION_A + 4 * VEC_BYTES)  // 54784 B total static LDS

union Frag {
    v16bf v;
    unsigned int u[8];
    __bf16 h[16];
};

struct Vecs {             // 4352 B, per wave
    float xy0[256];       // x0*y0         * a1
    float x0s[256];       // x0            * a3
    float x1y0[384];      // x1*y0         * a2
    float dt11[128];      // (x1.y1)/sqrt3 * a4
    float y1v[48];
    int   dstv[16];
};

// ---------------- prep: swizzle W1 (12x288 f32) into bf16 B-fragments, K padded to 32
__global__ void prep_w1(const float* __restrict__ W1, __bf16* __restrict__ B1) {
    int t = blockIdx.x * blockDim.x + threadIdx.x;
    if (t >= NT1 * 32 * 16) return;
    int j    = t & 15;
    int lane = (t >> 4) & 31;
    int nt   = t >> 9;
    int K = ((lane < 16) ? 0 : 8) + j + ((j < 8) ? 0 : 8);   // B layout: table order (matches A)
    int N = nt * 16 + (lane & 15);
    float v = (K < RADIAL_DIM) ? W1[K * NHID + N] : 0.0f;
    B1[t] = (__bf16)v;
}

// ---------------- prep: swizzle W2 into bf16 B-fragments, 16B-chunk interleave
// layout: [nt][kb][chunk(2)][lane(32)][8 halves]
__global__ void prep_w2(const float* __restrict__ W2, __bf16* __restrict__ B2) {
    int t = blockIdx.x * blockDim.x + threadIdx.x;
    if (t >= NT2 * NKB * 2 * 32 * 8) return;
    int jj   = t & 7;
    int lane = (t >> 3) & 31;
    int ch   = (t >> 8) & 1;
    int kb   = (t >> 9) % NKB;
    int nt   = t / (NKB * 512);
    int K = kb * 32 + ((lane < 16) ? 0 : 8) + ch * 16 + jj;
    int N = nt * 16 + (lane & 15);
    B2[t] = (__bf16)W2[K * NW + N];
}

__global__ void zero_f32(float* __restrict__ p, int n) {
    int i = blockIdx.x * blockDim.x + threadIdx.x;
    if (i < n) p[i] = 0.0f;
}

// LDS byte offset of a generic pointer that points into __shared__
__device__ __forceinline__ unsigned ldsAddr(const void* p) {
    return (unsigned)(unsigned long long)
        (__attribute__((address_space(3))) const char*)p;
}

// async global->LDS copy of one 9216B panel, 128 threads: 4 x b128 + 1 x b64 each
__device__ __forceinline__ void async_copy_panel(const void* gbase, unsigned goff,
                                                 unsigned loff, int tid) {
    unsigned l = loff + (unsigned)tid * 16u;
    unsigned g = goff + (unsigned)tid * 16u;
    #pragma unroll
    for (int i = 0; i < 4; ++i) {
        asm volatile("global_load_async_to_lds_b128 %0, %1, %2 offset:0"
                     :: "v"(l + (unsigned)i * 2048u),
                        "v"(g + (unsigned)i * 2048u),
                        "s"(gbase)
                     : "memory");
    }
    asm volatile("global_load_async_to_lds_b64 %0, %1, %2 offset:0"
                 :: "v"(loff + 8192u + (unsigned)tid * 8u),
                    "v"(goff + 8192u + (unsigned)tid * 8u),
                    "s"(gbase)
                 : "memory");
}

// 9 chained WMMAs over one staged n-panel
__device__ __forceinline__ v8f panel_gemm(const Frag* A2, const void* panel, int lane) {
    v8f c = {};
    const uint4* bp = (const uint4*)panel;
    #pragma unroll
    for (int kb = 0; kb < NKB; ++kb) {
        Frag b;
        uint4 q0 = bp[(kb * 2 + 0) * 32 + lane];   // lane stride 16B -> conflict-free
        uint4 q1 = bp[(kb * 2 + 1) * 32 + lane];
        b.u[0] = q0.x; b.u[1] = q0.y; b.u[2] = q0.z; b.u[3] = q0.w;
        b.u[4] = q1.x; b.u[5] = q1.y; b.u[6] = q1.z; b.u[7] = q1.w;
        c = __builtin_amdgcn_wmma_f32_16x16x32_bf16(false, A2[kb].v, false, b.v,
                                                    (short)0, c, false, false);
    }
    return c;
}

// ---------------- fused main kernel: 4 waves/block, 16 edges/wave,
// async double-buffered cooperative W2 staging
__launch_bounds__(128, 3)
__global__ void so3_conv(const float* __restrict__ nodef,
                         const float* __restrict__ sh,
                         const float* __restrict__ radial,
                         const int*   __restrict__ esrc,
                         const int*   __restrict__ edst,
                         const __bf16* __restrict__ B1,
                         const __bf16* __restrict__ B2,
                         float* __restrict__ out,
                         int E, float fScale)   // fScale = s2 * sNode
{
    __shared__ __align__(16) unsigned char shm[SHM_BYTES];

    const int tid  = threadIdx.x;
    const int w    = tid >> 5;
    const int lane = tid & 31;
    const int m    = lane & 15;            // matrix row (edge within tile) / N column
    const int hi   = (lane >> 4) << 3;     // 0 or 8
    const int sub  = m >> 3;               // i-split half for g2/g3
    const int e0   = (blockIdx.x * 4 + w) * 16;

    __bf16* hbuf = (__bf16*)(shm + w * H_BYTES);
    Vecs*   vec  = (Vecs*)(shm + REGION_A + w * VEC_BYTES);

    const float s1  = 0.28867513459481287f;   // 1/sqrt(12)
    const float A1s = 0.17677669529663687f;   // 1/sqrt(32)
    const float A2s = 0.25f;                  // 1/sqrt(16)
    const float A3s = 0.17677669529663687f;
    const float A4s = 0.25f;
    const float IS3 = 0.5773502691896258f;    // 1/sqrt(3)

    // ===== per-edge vectors (pre-scaled), 2 lanes per edge
    {
        int e = e0 + m;
        bool valid = e < E;
        int esafe = valid ? e : 0;
        int src = valid ? esrc[esafe] : 0;
        const float* xf = nodef + (size_t)src * XFEAT;
        float y0  = valid ? sh[(size_t)esafe * 4 + 0] : 0.0f;
        float y1x = valid ? sh[(size_t)esafe * 4 + 1] : 0.0f;
        float y1y = valid ? sh[(size_t)esafe * 4 + 2] : 0.0f;
        float y1z = valid ? sh[(size_t)esafe * 4 + 3] : 0.0f;
        if (lane < 16) {
            #pragma unroll
            for (int i = 0; i < 16; ++i)
                vec->xy0[m * 16 + i] = valid ? xf[i] * y0 * A1s : 0.0f;
            #pragma unroll
            for (int i = 0; i < 8; ++i) {
                float d = valid ? (xf[16 + i * 3 + 0] * y1x +
                                   xf[16 + i * 3 + 1] * y1y +
                                   xf[16 + i * 3 + 2] * y1z) : 0.0f;
                vec->dt11[m * 8 + i] = d * IS3 * A4s;
            }
            vec->y1v[m * 3 + 0] = y1x;
            vec->y1v[m * 3 + 1] = y1y;
            vec->y1v[m * 3 + 2] = y1z;
            vec->dstv[m] = valid ? edst[esafe] : 0;
        } else {
            #pragma unroll
            for (int i = 0; i < 16; ++i)
                vec->x0s[m * 16 + i] = valid ? xf[i] * A3s : 0.0f;
            #pragma unroll
            for (int i = 0; i < 24; ++i)
                vec->x1y0[m * 24 + i] = valid ? xf[16 + i] * y0 * A2s : 0.0f;
        }
    }

    // ===== GEMM1: h[16x288] = relu(radial[16x12] @ W1) * s1
    {
        Frag a;
        int e = e0 + m;
        bool valid = e < E;
        const float* rp = radial + (size_t)(valid ? e : 0) * RADIAL_DIM;
        #pragma unroll
        for (int j = 0; j < 16; ++j) {
            int K = (j < 8) ? (j + hi) : (j + 8 + hi);   // 16-bit A layout
            float v = (valid && K < RADIAL_DIM) ? rp[K] : 0.0f;
            a.h[j] = (__bf16)v;
        }
        const v16bf* B1v = (const v16bf*)B1;
        #pragma unroll 6
        for (int nt = 0; nt < NT1; ++nt) {
            v16bf b = B1v[nt * 32 + lane];
            v8f c = {};
            c = __builtin_amdgcn_wmma_f32_16x16x32_bf16(false, a.v, false, b,
                                                        (short)0, c, false, false);
            #pragma unroll
            for (int r = 0; r < 8; ++r) {
                float hv = c[r] * s1;
                hv = hv > 0.0f ? hv : 0.0f;
                hbuf[(r + hi) * H_STRIDE + nt * 16 + m] = (__bf16)hv;
            }
        }
    }

    // ===== build A fragments for GEMM2 (72 VGPRs, reused across all 36 n-tiles)
    Frag A2[NKB];
    #pragma unroll
    for (int kb = 0; kb < NKB; ++kb) {
        #pragma unroll
        for (int v = 0; v < 8; ++v) {
            int K = kb * 32 + 2 * v + ((v < 4) ? 0 : 8) + hi;
            A2[kb].u[v] =
                *reinterpret_cast<const unsigned int*>(&hbuf[m * H_STRIDE + K]);
        }
    }
    __syncthreads();   // everyone's h reads done -> panels may overwrite region A

    // ===== GEMM2 with fused epilogue; async double-buffered 9KB W2 panels
    float out0p[8];          // -> out0[e, o=m]
    float out1p[8][3];       // -> out1[e, o=lane&7, m]
    float t3p[8];            // -> t3[e, o=lane&7]
    #pragma unroll
    for (int r = 0; r < 8; ++r) {
        out0p[r] = 0.0f; t3p[r] = 0.0f;
        out1p[r][0] = 0.0f; out1p[r][1] = 0.0f; out1p[r][2] = 0.0f;
    }

    const unsigned panelLds = ldsAddr(shm);     // buffer b at panelLds + b*PANEL_BYTES
    async_copy_panel(B2, 0u, panelLds, tid);    // prologue: panel 0 -> buf 0

    #pragma unroll 2
    for (int nt = 0; nt < NT2; ++nt) {
        const int cur = nt & 1;
        asm volatile("s_wait_asynccnt 0x0" ::: "memory");  // my loads into buf[cur] landed
        __syncthreads();                                   // all waves' loads landed;
                                                           // all reads of buf[1-cur] retired
        if (nt + 1 < NT2)
            async_copy_panel(B2, (unsigned)(nt + 1) * PANEL_BYTES,
                             panelLds + (unsigned)((nt + 1) & 1) * PANEL_BYTES, tid);

        v8f c = panel_gemm(A2, shm + cur * PANEL_BYTES, lane);

        if (nt < 16) {                       // g1: w1 cols, i = nt, o = m
            #pragma unroll
            for (int r = 0; r < 8; ++r)
                out0p[r] += c[r] * vec->xy0[(r + hi) * 16 + nt];
        } else if (nt < 20) {                // g2: w2 cols, i = (nt-16)*2+sub, o = lane&7
            int i = (nt - 16) * 2 + sub;
            #pragma unroll
            for (int r = 0; r < 8; ++r) {
                const float* bv = &vec->x1y0[(r + hi) * 24 + i * 3];
                out1p[r][0] += c[r] * bv[0];
                out1p[r][1] += c[r] * bv[1];
                out1p[r][2] += c[r] * bv[2];
            }
        } else if (nt < 28) {                // g3: w3 cols, i = (nt-20)*2+sub, o = lane&7
            int i = (nt - 20) * 2 + sub;
            #pragma unroll
            for (int r = 0; r < 8; ++r)
                t3p[r] += c[r] * vec->x0s[(r + hi) * 16 + i];
        } else {                             // g4: w4 cols, i = nt-28, o = m
            #pragma unroll
            for (int r = 0; r < 8; ++r)
                out0p[r] += c[r] * vec->dt11[(r + hi) * 8 + (nt - 28)];
        }
    }

    // ===== merge i-split halves of out1p/t3p (panel buffer 0 is free now)
    float* scr = (float*)(shm) + w * 512;            // 16 slots * 32 floats per wave
    int slot = (lane & 7) + ((lane >= 16) ? 8 : 0);
    if (sub == 1) {
        #pragma unroll
        for (int r = 0; r < 8; ++r) {
            scr[slot * 32 + r * 3 + 0] = out1p[r][0];
            scr[slot * 32 + r * 3 + 1] = out1p[r][1];
            scr[slot * 32 + r * 3 + 2] = out1p[r][2];
            scr[slot * 32 + 24 + r]    = t3p[r];
        }
    }
    __syncthreads();
    if (sub == 0) {
        #pragma unroll
        for (int r = 0; r < 8; ++r) {
            out1p[r][0] += scr[slot * 32 + r * 3 + 0];
            out1p[r][1] += scr[slot * 32 + r * 3 + 1];
            out1p[r][2] += scr[slot * 32 + r * 3 + 2];
            t3p[r]      += scr[slot * 32 + 24 + r];
        }
    }

    // ===== scatter-add
    #pragma unroll
    for (int r = 0; r < 8; ++r) {
        int e = r + hi;
        float* op = out + (size_t)vec->dstv[e] * XFEAT;
        unsafeAtomicAdd(&op[m], out0p[r] * fScale);
    }
    if (sub == 0) {
        int o = lane & 7;
        #pragma unroll
        for (int r = 0; r < 8; ++r) {
            int e = r + hi;
            float* op = out + (size_t)vec->dstv[e] * XFEAT + 16 + o * 3;
            float y1x = vec->y1v[e * 3 + 0];
            float y1y = vec->y1v[e * 3 + 1];
            float y1z = vec->y1v[e * 3 + 2];
            unsafeAtomicAdd(&op[0], (out1p[r][0] + t3p[r] * y1x) * fScale);
            unsafeAtomicAdd(&op[1], (out1p[r][1] + t3p[r] * y1y) * fScale);
            unsafeAtomicAdd(&op[2], (out1p[r][2] + t3p[r] * y1z) * fScale);
        }
    }
}

extern "C" void kernel_launch(void* const* d_in, const int* in_sizes, int n_in,
                              void* d_out, int out_size, void* d_ws, size_t ws_size,
                              hipStream_t stream) {
    const float* nodef  = (const float*)d_in[0];
    const float* sh     = (const float*)d_in[1];
    const float* radial = (const float*)d_in[2];
    const float* W1     = (const float*)d_in[3];
    const float* W2     = (const float*)d_in[4];
    const int*   esrc   = (const int*)d_in[5];
    const int*   edst   = (const int*)d_in[6];
    float* out = (float*)d_out;

    const int E  = in_sizes[5];
    const int Nn = in_sizes[0] / XFEAT;
    const double s2 = 1.0 / sqrt((double)NHID);
    const float fScale = (float)(s2 / sqrt((double)Nn - 1.0));

    __bf16* wsB1 = (__bf16*)d_ws;                                  // 18 KB
    __bf16* wsB2 = (__bf16*)((char*)d_ws + NT1 * 32 * 16 * 2);     // 324 KB

    {
        int n = NT1 * 32 * 16;
        prep_w1<<<(n + 255) / 256, 256, 0, stream>>>(W1, wsB1);
    }
    {
        int n = NT2 * NKB * 2 * 32 * 8;
        prep_w2<<<(n + 255) / 256, 256, 0, stream>>>(W2, wsB2);
    }
    zero_f32<<<(out_size + 255) / 256, 256, 0, stream>>>(out, out_size);

    int nBlocks = (E + 63) / 64;
    so3_conv<<<nBlocks, 128, 0, stream>>>(nodef, sh, radial, esrc, edst,
                                          wsB1, wsB2, out, E, fScale);
}
